// MultiHeadAttention_40853728920117
// MI455X (gfx1250) — matrix operand order repaired and checked by reading, not compiled, not run on hardware
//
#include <hip/hip_runtime.h>
#include <hip/hip_bf16.h>
#include <cstdint>

#define DEVI __device__ __forceinline__

constexpr int N_ = 2;
constexpr int L_ = 2048;
constexpr int H_ = 16;
constexpr int D_ = 64;
constexpr int E_ = 1024;

typedef __attribute__((ext_vector_type(8)))  __bf16 v8bf;
typedef __attribute__((ext_vector_type(16))) __bf16 v16bf;
typedef __attribute__((ext_vector_type(8)))  float  v8f;

// gfx1250 async global->LDS path (guarded: falls back to synchronous copy)
#if __has_builtin(__builtin_amdgcn_global_load_async_to_lds_b128) && \
    __has_builtin(__builtin_amdgcn_s_wait_asynccnt)
#define USE_ASYNC_LDS 1
#else
#define USE_ASYNC_LDS 0
#endif

// builtin prototype is (v4i addrspace(1)*, v4i addrspace(3)*, Imm int, Imm int)
typedef __attribute__((ext_vector_type(4))) int i32x4;
typedef __attribute__((address_space(1))) i32x4 glb_i32x4;
typedef __attribute__((address_space(3))) i32x4 lds_i32x4;

DEVI void async_cp16(const __bf16* src, __bf16* dst_lds) {
#if USE_ASYNC_LDS
  __builtin_amdgcn_global_load_async_to_lds_b128((glb_i32x4*)src,
                                                 (lds_i32x4*)dst_lds, 0, 0);
#else
  *(v8bf*)dst_lds = *(const v8bf*)src;
#endif
}

DEVI void wait_async0() {
#if USE_ASYNC_LDS
  __builtin_amdgcn_s_wait_asynccnt(0);
#endif
}

DEVI void wait_async2() {
#if USE_ASYNC_LDS
  __builtin_amdgcn_s_wait_asynccnt(2);
#endif
}

DEVI v16bf cat8(v8bf lo, v8bf hi) {
  v16bf r;
#pragma unroll
  for (int i = 0; i < 8; ++i) { r[i] = lo[i]; r[i + 8] = hi[i]; }
  return r;
}

DEVI v8f zero8() {
  v8f z;
#pragma unroll
  for (int i = 0; i < 8; ++i) z[i] = 0.0f;
  return z;
}

DEVI v8f wmma_bf16(v16bf a, v16bf b, v8f c) {
  return __builtin_amdgcn_wmma_f32_16x16x32_bf16(false, a, false, b, (short)0, c,
                                                 false, false);
}

DEVI void put4(v16bf& v, int base, float4 f, float s) {
  v[base + 0] = (__bf16)(f.x * s);
  v[base + 1] = (__bf16)(f.y * s);
  v[base + 2] = (__bf16)(f.z * s);
  v[base + 3] = (__bf16)(f.w * s);
}

// ---------------------------------------------------------------------------
// Kernel 1: pack int32 mask (N,1,L,L) into 1 bit per entry.
// ---------------------------------------------------------------------------
__global__ void pack_mask_kernel(const int* __restrict__ mask,
                                 uint32_t* __restrict__ mbits) {
  int w = blockIdx.x * blockDim.x + threadIdx.x;  // one 32-entry word per thread
  const int* p = mask + (size_t)w * 32;
  uint32_t bits = 0;
#pragma unroll
  for (int j = 0; j < 32; ++j) bits |= (p[j] != 0 ? 1u : 0u) << j;
  mbits[w] = bits;
}

// ---------------------------------------------------------------------------
// Kernel 2: Wo fp32 -> bf16 (done once, reused by every output tile).
// ---------------------------------------------------------------------------
__global__ void cvt_bf16x4_kernel(const float* __restrict__ src,
                                  __bf16* __restrict__ dst) {
  int i = (blockIdx.x * blockDim.x + threadIdx.x) * 4;
  float4 f = *(const float4*)(src + i);
  dst[i + 0] = (__bf16)f.x;
  dst[i + 1] = (__bf16)f.y;
  dst[i + 2] = (__bf16)f.z;
  dst[i + 3] = (__bf16)f.w;
}

// ---------------------------------------------------------------------------
// Kernel 3: per-head projections via WMMA.
//   z==0: Qp[n,h,l,d] = (q @ Wq^T) * 1/sqrt(D)   (bf16)
//   z==1: Kp[n,h,l,d] =  k @ Wk^T                (bf16)
//   z==2: Vt[n,h,d,l] = (v @ Wv^T)^T             (bf16, transposed for PV B-frags)
// Block = 128 threads = 4 waves; each wave computes a 16x64 output tile.
// ---------------------------------------------------------------------------
__global__ __launch_bounds__(128) void project_kernel(
    const float* __restrict__ query, const float* __restrict__ keys,
    const float* __restrict__ values, const float* __restrict__ Wq,
    const float* __restrict__ Wk, const float* __restrict__ Wv,
    __bf16* __restrict__ Qp, __bf16* __restrict__ Kp, __bf16* __restrict__ Vt) {
  const int lane  = threadIdx.x & 31;
  const int wv    = threadIdx.x >> 5;
  const int col0  = lane & 15;
  const int half8 = (lane < 16) ? 0 : 8;
  const int aoff  = (lane < 16) ? 0 : 16;
  const int rbase = (blockIdx.x * 4 + wv) * 16;
  const int nh = blockIdx.y;
  const int n = nh >> 4, h = nh & 15;

  const float* src; const float* W; __bf16* dst; int transposed; float scale;
  if (blockIdx.z == 0)      { src = query;  W = Wq; dst = Qp; transposed = 0; scale = 0.125f; }
  else if (blockIdx.z == 1) { src = keys;   W = Wk; dst = Kp; transposed = 0; scale = 1.0f; }
  else                      { src = values; W = Wv; dst = Vt; transposed = 1; scale = 1.0f; }

  const int arow = rbase + col0;
  const float* ap = src + ((size_t)(n * L_ + arow)) * E_ + h * D_;

  v8f acc[4];
#pragma unroll
  for (int nt = 0; nt < 4; ++nt) acc[nt] = zero8();

#pragma unroll
  for (int c = 0; c < 2; ++c) {
    v16bf a;
    {
      const float* p = ap + c * 32 + aoff;
      put4(a, 0,  *(const float4*)(p + 0),  scale);
      put4(a, 4,  *(const float4*)(p + 4),  scale);
      put4(a, 8,  *(const float4*)(p + 8), scale);
      put4(a, 12, *(const float4*)(p + 12), scale);
    }
#pragma unroll
    for (int nt = 0; nt < 4; ++nt) {
      const int colW = nt * 16 + col0;
      const int koff = c * 32 + ((lane < 16) ? 0 : 16);
      const float* wp = W + colW * D_ + koff;
      v16bf b;
      put4(b, 0,  *(const float4*)(wp + 0),  1.0f);
      put4(b, 4,  *(const float4*)(wp + 4),  1.0f);
      put4(b, 8,  *(const float4*)(wp + 8),  1.0f);
      put4(b, 12, *(const float4*)(wp + 12), 1.0f);
      acc[nt] = wmma_bf16(a, b, acc[nt]);
    }
  }

#pragma unroll
  for (int nt = 0; nt < 4; ++nt) {
#pragma unroll
    for (int r = 0; r < 8; ++r) {
      const int row = rbase + r + half8;
      const int col = nt * 16 + col0;
      const __bf16 v = (__bf16)acc[nt][r];
      if (!transposed) dst[((size_t)nh * L_ + row) * D_ + col] = v;
      else             dst[((size_t)nh * D_ + col) * L_ + row] = v;
    }
  }
}

// ---------------------------------------------------------------------------
// Kernel 4: flash attention. 256 threads = 8 waves; each wave owns 16 q rows.
// K/V tiles for each 32-key chunk are staged once per block into LDS via the
// gfx1250 async global->LDS path, triple-buffered so a single s_barrier per
// chunk closes both the RAW and WAR windows. WMMA fragments are then fed from
// LDS (ds_load_b128), eliminating the 8x redundant per-wave global loads.
// ---------------------------------------------------------------------------
__global__ __launch_bounds__(256) void attn_kernel(
    const __bf16* __restrict__ Qp, const __bf16* __restrict__ Kp,
    const __bf16* __restrict__ Vt, const uint32_t* __restrict__ mbits,
    __bf16* __restrict__ Xattn) {
  __shared__ alignas(16) __bf16 plds[8 * 512];       // 8 KB: per-wave P staging
  __shared__ alignas(16) __bf16 ktile[3][32 * D_];   // 3 x 4 KB: K  [j][d]
  __shared__ alignas(16) __bf16 vtile[3][D_ * 32];   // 3 x 4 KB: V^T[d][j]
  const int lane  = threadIdx.x & 31;
  const int wv    = threadIdx.x >> 5;
  const int col0  = lane & 15;
  const int half8 = (lane < 16) ? 0 : 8;
  const int aoff  = (lane < 16) ? 0 : 16;
  const int koffB = (lane < 16) ? 0 : 16;
  const int qbase = blockIdx.x * 128 + wv * 16;
  const int nh = blockIdx.y;
  const int n = nh >> 4, h = nh & 15;
  __bf16* pw = plds + wv * 512;

  const __bf16* kg = Kp + (size_t)nh * L_ * D_;
  const __bf16* vg = Vt + (size_t)nh * D_ * L_;
  const int cid = wv * 32 + lane;        // 0..255: this thread's 16B copy slot
  const int vd = cid >> 2, vq = cid & 3; // V tile: 4 x 16B segments per d-row

  // Q A-fragments (1/sqrt(D) already folded into Qp)
  const __bf16* qrow = Qp + ((size_t)nh * L_ + (qbase + col0)) * D_;
  const v16bf a0 = cat8(*(const v8bf*)(qrow + aoff),
                        *(const v8bf*)(qrow + aoff + 8));
  const v16bf a1 = cat8(*(const v8bf*)(qrow + 32 + aoff),
                        *(const v8bf*)(qrow + 32 + aoff + 8));

  v8f o0 = zero8(), o1 = zero8(), o2 = zero8(), o3 = zero8();
  float m[8], l[8];
#pragma unroll
  for (int r = 0; r < 8; ++r) { m[r] = -1.0e30f; l[r] = 0.0f; }

  const size_t mwb = ((size_t)n * L_ + qbase) * (L_ / 32);
  constexpr float MASK_NEG = -1.0e30f;

  // prologue: fill buffer 0 (2 async-LDS instructions per wave)
  async_cp16(kg + (size_t)cid * 8, &ktile[0][cid * 8]);
  async_cp16(vg + (size_t)vd * L_ + vq * 8, &vtile[0][vd * 32 + vq * 8]);

  for (int jb = 0, it = 0; jb < L_; jb += 32, ++it) {
    const int cur = it % 3;
    if (jb + 32 < L_) {  // issue next chunk into buf (it+1)%3, keep 2 in flight
      const int nxt = (it + 1) % 3;
      async_cp16(kg + (size_t)(jb + 32) * D_ + cid * 8, &ktile[nxt][cid * 8]);
      async_cp16(vg + (size_t)vd * L_ + (jb + 32) + vq * 8,
                 &vtile[nxt][vd * 32 + vq * 8]);
      wait_async2();
    } else {
      wait_async0();
    }
    __syncthreads();  // buf[cur] fully resident for all 8 waves

    const __bf16* kl = ktile[cur];
    const __bf16* vl = vtile[cur];

    // S = Q x K^T   (two j-tiles x two d-chunks), fragments from LDS
    const __bf16* kp0 = kl + (size_t)col0 * D_ + koffB;
    const __bf16* kp1 = kl + (size_t)(16 + col0) * D_ + koffB;
    v8f s0 = wmma_bf16(a0, *(const v16bf*)(kp0), zero8());
    s0     = wmma_bf16(a1, *(const v16bf*)(kp0 + 32), s0);
    v8f s1 = wmma_bf16(a0, *(const v16bf*)(kp1), zero8());
    s1     = wmma_bf16(a1, *(const v16bf*)(kp1 + 32), s1);

    // mask + online softmax (row r lives in lanes 0-15 / 16-31 halves)
    const int jw = jb >> 5;
#pragma unroll
    for (int r = 0; r < 8; ++r) {
      const uint32_t word = mbits[mwb + (size_t)(r + half8) * (L_ / 32) + jw];
      if (!((word >> col0) & 1u))        s0[r] = MASK_NEG;
      if (!((word >> (col0 + 16)) & 1u)) s1[r] = MASK_NEG;

      float rmax = fmaxf(s0[r], s1[r]);
#pragma unroll
      for (int t = 1; t < 16; t <<= 1) rmax = fmaxf(rmax, __shfl_xor(rmax, t, 32));
      const float mnew = fmaxf(m[r], rmax);
      const float corr = __expf(m[r] - mnew);
      const float p0 = __expf(s0[r] - mnew);
      const float p1 = __expf(s1[r] - mnew);
      float rsum = p0 + p1;
#pragma unroll
      for (int t = 1; t < 16; t <<= 1) rsum += __shfl_xor(rsum, t, 32);
      m[r] = mnew;
      l[r] = l[r] * corr + rsum;
      o0[r] *= corr; o1[r] *= corr; o2[r] *= corr; o3[r] *= corr;
      pw[(r + half8) * 32 + col0]      = (__bf16)p0;
      pw[(r + half8) * 32 + col0 + 16] = (__bf16)p1;
    }

    // P: C-layout -> A-layout via per-wave LDS tile (intra-wave, DScnt-ordered)
    const __bf16* pr = pw + col0 * 32;
    const v16bf pa = cat8(*(const v8bf*)(pr + aoff),
                          *(const v8bf*)(pr + aoff + 8));

    // O += P x V  (V^T tile in LDS: column of V is contiguous)
    o0 = wmma_bf16(pa, *(const v16bf*)(vl + (size_t)(col0) * 32 + koffB), o0);
    o1 = wmma_bf16(pa, *(const v16bf*)(vl + (size_t)(16 + col0) * 32 + koffB), o1);
    o2 = wmma_bf16(pa, *(const v16bf*)(vl + (size_t)(32 + col0) * 32 + koffB), o2);
    o3 = wmma_bf16(pa, *(const v16bf*)(vl + (size_t)(48 + col0) * 32 + koffB), o3);
  }

  // normalize and store as bf16 in [n, l, h*64+d] layout for the output GEMM
  const size_t obase = ((size_t)n * L_ + qbase) * E_ + h * D_;
#pragma unroll
  for (int r = 0; r < 8; ++r) {
    const float inv = 1.0f / l[r];
    __bf16* op = Xattn + obase + (size_t)(r + half8) * E_;
    op[col0]      = (__bf16)(o0[r] * inv);
    op[16 + col0] = (__bf16)(o1[r] * inv);
    op[32 + col0] = (__bf16)(o2[r] * inv);
    op[48 + col0] = (__bf16)(o3[r] * inv);
  }
}

// ---------------------------------------------------------------------------
// Kernel 5: Y = X @ Wo^T + bo, X = (N*L, E) bf16, Wo bf16, fp32 out.
// 256 threads = 8 waves; each wave computes a 16x64 fp32 tile over K=1024.
// ---------------------------------------------------------------------------
__global__ __launch_bounds__(256) void out_gemm_kernel(
    const __bf16* __restrict__ X, const __bf16* __restrict__ Wob,
    const float* __restrict__ bo, float* __restrict__ out) {
  const int lane  = threadIdx.x & 31;
  const int wv    = threadIdx.x >> 5;
  const int col0  = lane & 15;
  const int half8 = (lane < 16) ? 0 : 8;
  const int aoff  = (lane < 16) ? 0 : 16;
  const int koffB = (lane < 16) ? 0 : 16;
  const int rbase = blockIdx.x * 128 + wv * 16;
  const int og = blockIdx.y * 64;

  v8f acc[4];
#pragma unroll
  for (int nt = 0; nt < 4; ++nt) acc[nt] = zero8();

  const __bf16* xr = X + (size_t)(rbase + col0) * E_;
  for (int kc = 0; kc < E_; kc += 32) {
    const v16bf a = cat8(*(const v8bf*)(xr + kc + aoff),
                         *(const v8bf*)(xr + kc + aoff + 8));
#pragma unroll
    for (int nt = 0; nt < 4; ++nt) {
      const __bf16* wp = Wob + (size_t)(og + nt * 16 + col0) * E_ + kc + koffB;
      acc[nt] = wmma_bf16(a, *(const v16bf*)(wp), acc[nt]);
    }
  }
#pragma unroll
  for (int nt = 0; nt < 4; ++nt) {
    const int col = og + nt * 16 + col0;
    const float bias = bo[col];
#pragma unroll
    for (int r = 0; r < 8; ++r) {
      const int row = rbase + r + half8;
      out[(size_t)row * E_ + col] = acc[nt][r] + bias;
    }
  }
}

// ---------------------------------------------------------------------------
extern "C" void kernel_launch(void* const* d_in, const int* in_sizes, int n_in,
                              void* d_out, int out_size, void* d_ws,
                              size_t ws_size, hipStream_t stream) {
  (void)in_sizes; (void)n_in; (void)out_size; (void)ws_size;
  const float* values = (const float*)d_in[0];
  const float* keys   = (const float*)d_in[1];
  const float* query  = (const float*)d_in[2];
  const int*   mask   = (const int*)d_in[3];
  const float* Wv     = (const float*)d_in[4];
  const float* Wk     = (const float*)d_in[5];
  const float* Wq     = (const float*)d_in[6];
  const float* Wo     = (const float*)d_in[7];
  const float* bo     = (const float*)d_in[8];
  float* out = (float*)d_out;

  // workspace carve-up (~36.7 MB total)
  const size_t SZ = (size_t)N_ * H_ * L_ * D_;  // 4,194,304 elems
  __bf16* Qp    = (__bf16*)d_ws;
  __bf16* Kp    = Qp + SZ;
  __bf16* Vt    = Kp + SZ;
  __bf16* Xattn = Vt + SZ;
  __bf16* Wob   = Xattn + SZ;
  uint32_t* mbits = (uint32_t*)(Wob + (size_t)E_ * E_);

  pack_mask_kernel<<<dim3((N_ * L_ * L_ / 32) / 256), dim3(256), 0, stream>>>(
      mask, mbits);
  cvt_bf16x4_kernel<<<dim3((E_ * E_ / 4) / 256), dim3(256), 0, stream>>>(Wo, Wob);
  project_kernel<<<dim3(L_ / 64, N_ * H_, 3), dim3(128), 0, stream>>>(
      query, keys, values, Wq, Wk, Wv, Qp, Kp, Vt);
  attn_kernel<<<dim3(L_ / 128, N_ * H_), dim3(256), 0, stream>>>(
      Qp, Kp, Vt, mbits, Xattn);
  out_gemm_kernel<<<dim3(N_ * L_ / 128, E_ / 64), dim3(256), 0, stream>>>(
      Xattn, Wob, bo, out);
}
